// RNN_84043920048424
// MI455X (gfx1250) — matrix operand order repaired and checked by reading, not compiled, not run on hardware
//
#include <hip/hip_runtime.h>
#include <hip/hip_bf16.h>

#define VOCAB 50257
#define EMB   1024
#define HID   2048
#define SEQ   2048

typedef __attribute__((ext_vector_type(16))) __bf16 v16bf;
typedef __attribute__((ext_vector_type(8)))  __bf16 v8bf;
typedef __attribute__((ext_vector_type(8)))  float  v8f;

// ---------------- workspace layout (bytes) ----------------
// We_bf : HID x EMB bf16   (4 MB)
// Wh_bf : HID x HID bf16   (8 MB)
// E_bf  : SEQ x EMB bf16   (4 MB)
// Ut    : SEQ x HID f32    (16 MB)  Ut[t][h] = (W_e e_t + b)[h]  (row-major, stride HID)
// hbuf  : 2 x HID f32      (double-buffered hidden state)
// ctrl  : 2 x u32          (grid-barrier count / generation)
#define OFF_WE   ((size_t)0)
#define OFF_WH   (OFF_WE + (size_t)HID * EMB * 2)
#define OFF_EB   (OFF_WH + (size_t)HID * HID * 2)
#define OFF_U    (OFF_EB + (size_t)SEQ * EMB * 2)
#define OFF_H    (OFF_U  + (size_t)SEQ * HID * 4)
#define OFF_CTRL (OFF_H  + (size_t)2 * HID * 4)

#define RNN_BLOCKS 128
#define GEMM_WAVES 8

union Frag { v16bf v; v8bf h[2]; };

// -------- 1) gather embeddings -> bf16; split/convert W_i2h --------
__global__ void __launch_bounds__(256) rnn_prep_kernel(
    const int* __restrict__ tokens, const float* __restrict__ emb,
    const float* __restrict__ Wi2h,
    __bf16* __restrict__ We, __bf16* __restrict__ Wh, __bf16* __restrict__ Eb) {
  size_t idx    = (size_t)blockIdx.x * 256 + threadIdx.x;
  size_t stride = (size_t)gridDim.x * 256;

  const size_t wtot = (size_t)HID * (EMB + HID);
  for (size_t i = idx; i < wtot; i += stride) {
    size_t r = i / (EMB + HID);
    size_t c = i - r * (EMB + HID);
    float v = Wi2h[i];
    if (c < EMB) We[r * EMB + c] = static_cast<__bf16>(v);
    else         Wh[r * HID + (c - EMB)] = static_cast<__bf16>(v);
  }
  const size_t etot = (size_t)SEQ * EMB;
  for (size_t i = idx; i < etot; i += stride) {
    size_t t = i / EMB;
    size_t k = i - t * EMB;
    int tok = tokens[t];
    tok = tok < 0 ? 0 : (tok >= VOCAB ? VOCAB - 1 : tok);
    Eb[i] = static_cast<__bf16>(emb[(size_t)tok * EMB + k]);
  }
}

// fragment loader: A (16x32) + 4x B (32x16) for one K-step
__device__ __forceinline__ void load_frags(
    const __bf16* __restrict__ Abase, const __bf16* __restrict__ Bbase,
    int k0, Frag& a, Frag (&b)[4]) {
  a.h[0] = *(const v8bf*)(Abase + k0);
  a.h[1] = *(const v8bf*)(Abase + k0 + 16);
  #pragma unroll
  for (int s = 0; s < 4; ++s) {
    const __bf16* bp = Bbase + (size_t)(s * 16) * EMB + k0;
    b[s].h[0] = *(const v8bf*)(bp);
    b[s].h[1] = *(const v8bf*)(bp + 16);
  }
}

// -------- 2) Ut = E @ We^T + b  via bf16 WMMA (M=SEQ, N=HID, K=EMB) --------
// Each wave computes a 16x64 strip: one A (E) fragment reused across 4 B (We)
// tiles. Software-pipelined: iteration k+1's 10 b128 loads are issued before
// iteration k's WMMAs, so each WMMA waits only on already-old loads (partial
// loadcnt wait) while the next load clause is in flight.
__global__ void __launch_bounds__(256) rnn_gemm_u_kernel(
    const __bf16* __restrict__ Eb, const __bf16* __restrict__ We,
    const float* __restrict__ bias, float* __restrict__ Ut) {
  const int wave = threadIdx.x >> 5;
  const int lane = threadIdx.x & 31;
  const int job  = blockIdx.x * GEMM_WAVES + wave;   // 128 m-tiles x 32 n-strips
  const int m0 = (job >> 5) << 4;                    // t-tile base (16 rows)
  const int n0 = (job & 31) << 6;                    // h-strip base (64 cols)
  const int lr = lane & 15;
  const int lh = lane >> 4;

  // A 16x32 bf16: lane<16 -> K {0..7,16..23}, lane>=16 -> K {8..15,24..31}
  const __bf16* Abase = Eb + (size_t)(m0 + lr) * EMB + lh * 8;
  // B 32x16 bf16: lane = column; lane<16 -> K {0..7,16..23}, lane>=16 -> K {8..15,24..31}
  const __bf16* Bbase = We + (size_t)(n0 + lr) * EMB + lh * 8;

  v8f acc0 = {}, acc1 = {}, acc2 = {}, acc3 = {};
  Frag a, b[4], an, bn[4];
  load_frags(Abase, Bbase, 0, a, b);

  // Note: the last prefetch (k0 = EMB) reads one K-step past the tile; that
  // address is still inside the workspace (Eb is followed by Ut) and the data
  // is never consumed.
  #pragma unroll 2
  for (int k0 = 0; k0 < EMB; k0 += 32) {
    load_frags(Abase, Bbase, k0 + 32, an, bn);
    acc0 = __builtin_amdgcn_wmma_f32_16x16x32_bf16(false, a.v, false, b[0].v,
                                                   (short)0, acc0, false, false);
    acc1 = __builtin_amdgcn_wmma_f32_16x16x32_bf16(false, a.v, false, b[1].v,
                                                   (short)0, acc1, false, false);
    acc2 = __builtin_amdgcn_wmma_f32_16x16x32_bf16(false, a.v, false, b[2].v,
                                                   (short)0, acc2, false, false);
    acc3 = __builtin_amdgcn_wmma_f32_16x16x32_bf16(false, a.v, false, b[3].v,
                                                   (short)0, acc3, false, false);
    a = an;
    #pragma unroll
    for (int s = 0; s < 4; ++s) b[s] = bn[s];
  }

  // C/D layout: VGPR i -> M = i (lanes 0..15) or 8+i (lanes 16..31); N = lane&15
  const int trow = m0 + lh * 8;
  v8f accs[4] = {acc0, acc1, acc2, acc3};
  #pragma unroll
  for (int s = 0; s < 4; ++s) {
    const int col = n0 + s * 16 + lr;
    const float bv = bias[col];
    #pragma unroll
    for (int i = 0; i < 8; ++i)
      Ut[(size_t)(trow + i) * HID + col] = accs[s][i] + bv;
  }
}

// -------- 3) persistent sequential recurrence: h <- Wh*h + Ut[t,:] --------
// 128 blocks x 256 threads; 16 rows per block, 16 lanes per row
// (128 FMAs/lane/step; W_h streamed from L2 at full width).
__global__ void __launch_bounds__(256) rnn_recur_kernel(
    const __bf16* __restrict__ Wh, const float* __restrict__ Ut,
    float* __restrict__ hbuf, unsigned* __restrict__ ctrl) {
  __shared__ float sh[HID];
  const int tid  = threadIdx.x;
  const int part = tid & 15;                    // 1/16th-row worker
  const int row  = blockIdx.x * 16 + (tid >> 4);
  const __bf16* wrow = Wh + (size_t)row * HID + part * (HID / 16);
  unsigned* cnt = ctrl;
  unsigned* gen = ctrl + 1;

  for (int t = 0; t < SEQ; ++t) {
    // stage current h into LDS (broadcast to whole block)
    const float* hcur = hbuf + (size_t)(t & 1) * HID;
    #pragma unroll
    for (int i = 0; i < HID / 256; ++i)
      sh[tid + i * 256] = hcur[tid + i * 256];
    __syncthreads();

    // 1/16th-row dot product: 128 bf16 weights, streamed 16B at a time
    float accd = 0.0f;
    const float* hp = sh + part * (HID / 16);
    #pragma unroll
    for (int j = 0; j < HID / 16; j += 8) {
      v8bf w = *(const v8bf*)(wrow + j);
      #pragma unroll
      for (int q = 0; q < 8; ++q)
        accd = fmaf((float)w[q], hp[j + q], accd);
    }
    // combine the 16 partial sums (contiguous lanes of the same wave32)
    accd += __shfl_xor(accd, 1);
    accd += __shfl_xor(accd, 2);
    accd += __shfl_xor(accd, 4);
    accd += __shfl_xor(accd, 8);
    if (part == 0)
      hbuf[(size_t)((t + 1) & 1) * HID + row] = accd + Ut[(size_t)t * HID + row];

    // device-wide barrier between steps
    __syncthreads();
    if (tid == 0) {
      __threadfence();
      unsigned a = __hip_atomic_fetch_add(cnt, 1u, __ATOMIC_ACQ_REL,
                                          __HIP_MEMORY_SCOPE_AGENT);
      if (a == (unsigned)(RNN_BLOCKS - 1)) {
        __hip_atomic_store(cnt, 0u, __ATOMIC_RELAXED, __HIP_MEMORY_SCOPE_AGENT);
        __hip_atomic_fetch_add(gen, 1u, __ATOMIC_ACQ_REL,
                               __HIP_MEMORY_SCOPE_AGENT);
      } else {
        while (__hip_atomic_load(gen, __ATOMIC_ACQUIRE,
                                 __HIP_MEMORY_SCOPE_AGENT) < (unsigned)(t + 1)) {
          __builtin_amdgcn_s_sleep(1);
        }
      }
    }
    __syncthreads();
  }
}

// -------- 4) out = sigmoid(W_h2o . h_final + b_h2o) --------
__global__ void __launch_bounds__(256) rnn_out_kernel(
    const float* __restrict__ Wo, const float* __restrict__ bo,
    const float* __restrict__ hfin, float* __restrict__ out) {
  __shared__ float red[256];
  float acc = 0.0f;
  for (int i = threadIdx.x; i < HID; i += 256)
    acc = fmaf(Wo[i], hfin[i], acc);
  red[threadIdx.x] = acc;
  __syncthreads();
  #pragma unroll
  for (int s = 128; s > 0; s >>= 1) {
    if (threadIdx.x < s) red[threadIdx.x] += red[threadIdx.x + s];
    __syncthreads();
  }
  if (threadIdx.x == 0) {
    float z = red[0] + bo[0];
    out[0] = 1.0f / (1.0f + expf(-z));
  }
}

extern "C" void kernel_launch(void* const* d_in, const int* in_sizes, int n_in,
                              void* d_out, int out_size, void* d_ws, size_t ws_size,
                              hipStream_t stream) {
  const int*   tokens = (const int*)d_in[0];
  const float* emb    = (const float*)d_in[1];
  const float* Wi2h   = (const float*)d_in[2];
  const float* b_i2h  = (const float*)d_in[3];
  const float* Wh2o   = (const float*)d_in[4];
  const float* b_h2o  = (const float*)d_in[5];
  (void)in_sizes; (void)n_in; (void)out_size; (void)ws_size;

  char* ws = (char*)d_ws;
  __bf16*   We   = (__bf16*)(ws + OFF_WE);
  __bf16*   Wh   = (__bf16*)(ws + OFF_WH);
  __bf16*   Eb   = (__bf16*)(ws + OFF_EB);
  float*    Ut   = (float*)(ws + OFF_U);
  float*    hbuf = (float*)(ws + OFF_H);
  unsigned* ctrl = (unsigned*)(ws + OFF_CTRL);

  // zero hidden state (h0 = 0) and barrier control words (contiguous region)
  hipMemsetAsync(ws + OFF_H, 0, (size_t)2 * HID * 4 + 64, stream);

  rnn_prep_kernel<<<512, 256, 0, stream>>>(tokens, emb, Wi2h, We, Wh, Eb);

  // (SEQ/16) m-tiles x (HID/64) n-strips = 4096 wave jobs, 8 waves/block
  rnn_gemm_u_kernel<<<(128 * 32) / GEMM_WAVES, 256, 0, stream>>>(Eb, We, b_i2h, Ut);

  rnn_recur_kernel<<<RNN_BLOCKS, 256, 0, stream>>>(Wh, Ut, hbuf, ctrl);

  rnn_out_kernel<<<1, 256, 0, stream>>>(Wh2o, b_h2o, hbuf, (float*)d_out);
}
